// ContrastiveLoss_63625645523217
// MI455X (gfx1250) — matrix operand order repaired and checked by reading, not compiled, not run on hardware
//
#include <hip/hip_runtime.h>
#include <hip/hip_bf16.h>

typedef __attribute__((ext_vector_type(16))) _Float16 v16h;
typedef __attribute__((ext_vector_type(8)))  _Float16 v8h;
typedef __attribute__((ext_vector_type(4)))  _Float16 v4h;
typedef __attribute__((ext_vector_type(8)))  float    v8f;

#define B_N   4096
#define D_K   512
#define INV_T 10.0f          // 1 / 0.1
#define KCHUNKS (D_K / 32)   // 16 wmma steps over K
#define NTILES  (B_N / 16)   // 256 row/col tiles

// ---------------------------------------------------------------------------
// Fragment loaders in this file's own contiguous k order (k = 16*(lane/16) + s), same for A and B:
//  A 16x32 f16: lanes 0-15 hold row M=lane, K=0..15; lanes 16-31
//               hold row M=lane-16, K=16..31  -> two 16B chunks.
//  B 32x16 f16: lane holds column N=lane%16; lanes 0-15 K=0..15,
//               lanes 16-31 K=16..31 -> one contiguous 32B chunk of E's row.
// ---------------------------------------------------------------------------
__device__ __forceinline__ v16h load_a_frag_lds(const _Float16* sA, int kc, int lane) {
    int m  = lane & 15;
    int kb = kc * 32 + ((lane & 16) ? 16 : 0);
    const _Float16* p = sA + m * D_K + kb;
    v8h lo = *(const v8h*)(p);
    v8h hi = *(const v8h*)(p + 8);
    v16h a;
#pragma unroll
    for (int i = 0; i < 8; ++i) { a[i] = lo[i]; a[i + 8] = hi[i]; }
    return a;
}

__device__ __forceinline__ v16h load_b_frag(const _Float16* __restrict__ E16,
                                            int colBase, int kc, int lane) {
    int col = colBase + (lane & 15);
    int kb  = kc * 32 + ((lane & 16) ? 16 : 0);
    return *(const v16h*)(E16 + col * D_K + kb);   // 32B aligned
}

// Stage this block's 16x512 f16 row tile into LDS (16 KB) with b128 traffic.
__device__ __forceinline__ void stage_rows(_Float16* sA, const _Float16* __restrict__ E16,
                                           int rowBase, int tid) {
    const uint4* src = (const uint4*)(E16 + rowBase * D_K);
    uint4*       dst = (uint4*)sA;
#pragma unroll
    for (int i = tid; i < (16 * D_K * 2) / 16; i += 256)   // 1024 uint4
        dst[i] = src[i];
}

// ---------------------------------------------------------------------------
__global__ void cl_convert_f16(const float* __restrict__ in, _Float16* __restrict__ out) {
    int idx = (blockIdx.x * 256 + threadIdx.x) * 4;
    float4 v = *(const float4*)(in + idx);
    v4h h;
    h[0] = (_Float16)v.x; h[1] = (_Float16)v.y;
    h[2] = (_Float16)v.z; h[3] = (_Float16)v.w;
    *(v4h*)(out + idx) = h;
}

__global__ void cl_init(float* __restrict__ denom, float* __restrict__ scalars) {
    int i = blockIdx.x * 256 + threadIdx.x;
    if (i < B_N) denom[i] = 0.0f;
    if (i < 2)   scalars[i] = 0.0f;
}

// Pass 1: denom[r] = sum_{j : label[j]!=label[r]} exp(sims[r,j])
// Each wave runs two independent WMMA accumulator chains (tiles jt, jt+8)
// so dependent v_wmma ops from one chain hide in the XDL pipe behind the other.
__global__ void __launch_bounds__(256)
cl_pass1(const _Float16* __restrict__ E16, const int* __restrict__ labels,
         float* __restrict__ denom) {
    __shared__ __align__(32) _Float16 sA[16 * D_K];
    __shared__ float sDenom[16];

    const int tid  = threadIdx.x;
    const int lane = tid & 31;
    const int wave = tid >> 5;
    const int rowBase = blockIdx.x * 16;
    const int rowOff  = (lane & 16) ? 8 : 0;

    if (tid < 16) sDenom[tid] = 0.0f;
    stage_rows(sA, E16, rowBase, tid);
    __syncthreads();

    int rlab[8];
#pragma unroll
    for (int v = 0; v < 8; ++v) rlab[v] = labels[rowBase + rowOff + v];

    float dsum[8];
#pragma unroll
    for (int v = 0; v < 8; ++v) dsum[v] = 0.0f;

    for (int jt = wave; jt < NTILES; jt += 16) {
        const int colBase0 = jt * 16;
        const int colBase1 = (jt + 8) * 16;
        const int clab0 = labels[colBase0 + (lane & 15)];
        const int clab1 = labels[colBase1 + (lane & 15)];
        v8f acc0 = {};
        v8f acc1 = {};
#pragma unroll
        for (int kc = 0; kc < KCHUNKS; ++kc) {
            v16h a  = load_a_frag_lds(sA, kc, lane);
            v16h b0 = load_b_frag(E16, colBase0, kc, lane);
            v16h b1 = load_b_frag(E16, colBase1, kc, lane);
            acc0 = __builtin_amdgcn_wmma_f32_16x16x32_f16(
                false, a, false, b0, (short)0, acc0, false, false);
            acc1 = __builtin_amdgcn_wmma_f32_16x16x32_f16(
                false, a, false, b1, (short)0, acc1, false, false);
        }
#pragma unroll
        for (int v = 0; v < 8; ++v) {
            float e0 = __expf(acc0[v] * INV_T);
            float e1 = __expf(acc1[v] * INV_T);
            dsum[v] += (rlab[v] != clab0) ? e0 : 0.0f;   // negatives only
            dsum[v] += (rlab[v] != clab1) ? e1 : 0.0f;
        }
    }

    // Reduce over the 16 lanes that share each row (C/D layout: N = lane%16).
#pragma unroll
    for (int v = 0; v < 8; ++v) {
        float x = dsum[v];
        x += __shfl_xor(x, 1, 16);
        x += __shfl_xor(x, 2, 16);
        x += __shfl_xor(x, 4, 16);
        x += __shfl_xor(x, 8, 16);
        if ((lane & 15) == 0) atomicAdd(&sDenom[rowOff + v], x);  // ds_add_f32
    }
    __syncthreads();
    if (tid < 16) denom[rowBase + tid] = sDenom[tid];  // tile owned by one WG
}

// Pass 2: loss_sum += log(denom[r] + e) - s over positive pairs; count them.
__global__ void __launch_bounds__(256)
cl_pass2(const _Float16* __restrict__ E16, const int* __restrict__ labels,
         const float* __restrict__ denom, float* __restrict__ scalars) {
    __shared__ __align__(32) _Float16 sA[16 * D_K];
    __shared__ float sAcc[2];

    const int tid  = threadIdx.x;
    const int lane = tid & 31;
    const int wave = tid >> 5;
    const int rowBase = blockIdx.x * 16;
    const int rowOff  = (lane & 16) ? 8 : 0;

    if (tid < 2) sAcc[tid] = 0.0f;
    stage_rows(sA, E16, rowBase, tid);
    __syncthreads();

    int   rlab[8];
    float rden[8];
#pragma unroll
    for (int v = 0; v < 8; ++v) {
        int r   = rowBase + rowOff + v;
        rlab[v] = labels[r];
        rden[v] = denom[r];
    }

    float lsum = 0.0f, pcnt = 0.0f;

    for (int jt = wave; jt < NTILES; jt += 16) {
        const int colBase0 = jt * 16;
        const int colBase1 = (jt + 8) * 16;
        const int myCol0 = colBase0 + (lane & 15);
        const int myCol1 = colBase1 + (lane & 15);
        const int clab0  = labels[myCol0];
        const int clab1  = labels[myCol1];
        v8f acc0 = {};
        v8f acc1 = {};
#pragma unroll
        for (int kc = 0; kc < KCHUNKS; ++kc) {
            v16h a  = load_a_frag_lds(sA, kc, lane);
            v16h b0 = load_b_frag(E16, colBase0, kc, lane);
            v16h b1 = load_b_frag(E16, colBase1, kc, lane);
            acc0 = __builtin_amdgcn_wmma_f32_16x16x32_f16(
                false, a, false, b0, (short)0, acc0, false, false);
            acc1 = __builtin_amdgcn_wmma_f32_16x16x32_f16(
                false, a, false, b1, (short)0, acc1, false, false);
        }
#pragma unroll
        for (int v = 0; v < 8; ++v) {
            const int r = rowBase + rowOff + v;
            if ((rlab[v] == clab0) && (r != myCol0)) {     // positive, no diag
                float s = acc0[v] * INV_T;
                float e = __expf(s);
                lsum += __logf(rden[v] + e) - s;           // -log(e/(d+e))
                pcnt += 1.0f;
            }
            if ((rlab[v] == clab1) && (r != myCol1)) {
                float s = acc1[v] * INV_T;
                float e = __expf(s);
                lsum += __logf(rden[v] + e) - s;
                pcnt += 1.0f;
            }
        }
    }

    // full-wave butterfly, then LDS, then one global atomic per block
    for (int m = 16; m >= 1; m >>= 1) {
        lsum += __shfl_xor(lsum, m, 32);
        pcnt += __shfl_xor(pcnt, m, 32);
    }
    if (lane == 0) {
        atomicAdd(&sAcc[0], lsum);
        atomicAdd(&sAcc[1], pcnt);
    }
    __syncthreads();
    if (tid == 0) {
        atomicAdd(&scalars[0], sAcc[0]);
        atomicAdd(&scalars[1], sAcc[1]);
    }
}

__global__ void cl_finalize(const float* __restrict__ scalars, float* __restrict__ out) {
    if (threadIdx.x == 0 && blockIdx.x == 0)
        out[0] = scalars[0] / fmaxf(scalars[1], 1.0f);
}

// ---------------------------------------------------------------------------
extern "C" void kernel_launch(void* const* d_in, const int* in_sizes, int n_in,
                              void* d_out, int out_size, void* d_ws, size_t ws_size,
                              hipStream_t stream) {
    (void)in_sizes; (void)n_in; (void)out_size; (void)ws_size;
    const float* emb    = (const float*)d_in[0];
    const int*   labels = (const int*)d_in[1];
    float*       out    = (float*)d_out;

    char* ws = (char*)d_ws;
    _Float16* e16     = (_Float16*)ws;                                   // 4 MB
    float*    denom   = (float*)(ws + (size_t)B_N * D_K * sizeof(_Float16));
    float*    scalars = denom + B_N;                                     // [loss_sum, num_pos]

    cl_convert_f16<<<(B_N * D_K / 4) / 256, 256, 0, stream>>>(emb, e16);
    cl_init<<<(B_N + 255) / 256, 256, 0, stream>>>(denom, scalars);
    cl_pass1<<<NTILES, 256, 0, stream>>>(e16, labels, denom);
    cl_pass2<<<NTILES, 256, 0, stream>>>(e16, labels, denom, scalars);
    cl_finalize<<<1, 32, 0, stream>>>(scalars, out);
}
